// Attention_msa_41772851921145
// MI455X (gfx1250) — matrix operand order repaired and checked by reading, not compiled, not run on hardware
//
#include <hip/hip_runtime.h>
#include <hip/hip_bf16.h>

#define N_TOK 2048
#define C_IN  1024
#define M_OUT 3072
#define NH    8
#define HD    128
#define SCALE_F 25.0f

typedef __bf16 bf16;
typedef __attribute__((ext_vector_type(8)))  __bf16 v8bf;
typedef __attribute__((ext_vector_type(16))) __bf16 v16bf;
typedef __attribute__((ext_vector_type(8)))  float  v8f;

__device__ __forceinline__ v16bf bf_combine(v8bf lo, v8bf hi) {
  return __builtin_shufflevector(lo, hi, 0,1,2,3,4,5,6,7,8,9,10,11,12,13,14,15);
}

// A fragment 16x32 (MxK): lane&15 = M, lane>>4 picks K-half base {0,8};
// VGPR0-3 hold K kb..kb+7, VGPR4-7 hold K kb+16..kb+23.
__device__ __forceinline__ v16bf frag_a(const bf16* base, int stride, int lane) {
  const bf16* p = base + (lane & 15) * stride + ((lane >> 4) << 3);
  return bf_combine(*(const v8bf*)p, *(const v8bf*)(p + 16));
}

// B fragment 32x16 (KxN): lane&15 = N, lane>>4 picks K-half base {0,8};
// elements 0-7 hold K kb..kb+7, elements 8-15 hold K kb+16..kb+23.
__device__ __forceinline__ v16bf frag_b(const bf16* base, int stride, int lane) {
  const bf16* p = base + ((lane >> 4) << 3) * stride + (lane & 15);
  v16bf r;
#pragma unroll
  for (int s = 0; s < 8; ++s) {
    r[s] = p[s * stride];
    r[s + 8] = p[(s + 16) * stride];
  }
  return r;
}

__device__ __forceinline__ v8f wmma_bf16(v16bf a, v16bf b, v8f c) {
  return __builtin_amdgcn_wmma_f32_16x16x32_bf16(false, a, false, b, (short)0, c,
                                                 false, false);
}

// ---------------------------------------------------------------- converts
__global__ __launch_bounds__(256) void conv_x_kern(const float* __restrict__ x,
                                                   bf16* __restrict__ X16) {
  int i = (blockIdx.x * 256 + threadIdx.x) * 8;     // N_TOK*C_IN divisible by 2048
  float4 a = *(const float4*)(x + i);
  float4 b = *(const float4*)(x + i + 4);
  v8bf o;
  o[0] = (bf16)a.x; o[1] = (bf16)a.y; o[2] = (bf16)a.z; o[3] = (bf16)a.w;
  o[4] = (bf16)b.x; o[5] = (bf16)b.y; o[6] = (bf16)b.z; o[7] = (bf16)b.w;
  *(v8bf*)(X16 + i) = o;
}

// 64x64 LDS tile transpose: W[m][c] fp32 -> WT16[c][m] bf16, coalesced both sides.
__global__ __launch_bounds__(256) void conv_wt_kern(const float* __restrict__ W,
                                                    bf16* __restrict__ WT16) {
  __shared__ bf16 lT[64 * 72];          // [c][m], stride 72 (pad)
  const int tid = threadIdx.x;
  const int m0 = blockIdx.x * 64, c0 = blockIdx.y * 64;
  float4 ta[2][2];
#pragma unroll
  for (int u = 0; u < 2; ++u) {
    int v = tid + 256 * u;
    int mm = v >> 3, c8 = (v & 7) * 8;
    const float* src = W + (size_t)(m0 + mm) * C_IN + c0 + c8;
    ta[u][0] = *(const float4*)src;
    ta[u][1] = *(const float4*)(src + 4);
  }
#pragma unroll
  for (int u = 0; u < 2; ++u) {
    int v = tid + 256 * u;
    int mm = v >> 3, c8 = (v & 7) * 8;
    const float* s = (const float*)&ta[u][0];
#pragma unroll
    for (int e = 0; e < 8; ++e) lT[(c8 + e) * 72 + mm] = (bf16)s[e];
  }
  __syncthreads();
  v8bf to[2];
#pragma unroll
  for (int u = 0; u < 2; ++u) {
    int v = tid + 256 * u;
    int cc = v >> 3, m8 = (v & 7) * 8;
    to[u] = *(const v8bf*)&lT[cc * 72 + m8];
  }
#pragma unroll
  for (int u = 0; u < 2; ++u) {
    int v = tid + 256 * u;
    int cc = v >> 3, m8 = (v & 7) * 8;
    *(v8bf*)(WT16 + (size_t)(c0 + cc) * M_OUT + m0 + m8) = to[u];
  }
}

// ------------------------------------------------- QKV GEMM + cosine norms
// Y[n,m] = sum_c X[n,c]*W[m,c];  tile 128x128, K staged in 32-steps.
// Epilogue lays out tensors exactly how downstream WMMA passes consume them:
//   Q16  [h][n][d] row-major  (A operand of Q@K^T)
//   KT16 [h][d][n] transposed (B operand of Q@K^T)   - via LDS transpose
//   Vn16 [h][n][d] row-major  (A operand of vn@vn^T)
//   VnT16[h][d][n] transposed (B operand of vn@vn^T) - via LDS transpose
//   Vr16 [h][n][d] row-major  (B operand of P@V)
__global__ __launch_bounds__(256) void qkv_gemm_kern(
    const bf16* __restrict__ X16, const bf16* __restrict__ WT16,
    bf16* __restrict__ Q16, bf16* __restrict__ KT16, bf16* __restrict__ Vn16,
    bf16* __restrict__ VnT16, bf16* __restrict__ Vr16, float* __restrict__ out) {
  __shared__ bf16 lA[128 * 40];   // [row][k] stride 40 (pad)
  __shared__ bf16 lB[32 * 136];   // [k][n]  stride 136 (pad)
  __shared__ bf16 lT[128 * 136];  // [d][n]  epilogue transpose tile
  const int tid = threadIdx.x, lane = tid & 31, wave = tid >> 5;
  const int m0 = blockIdx.x * 128;
  const int nt = blockIdx.y;            // 0..23
  const int t = nt >> 3, h = nt & 7;    // qkv slot, head
  const int n0 = nt * 128;

  v8f acc[8] = {};
  for (int k0 = 0; k0 < C_IN; k0 += 32) {
    __syncthreads();
    {                                   // stage A(128x32) + B(32x128), MLP-friendly
      v8bf ta[2], tb[2];
#pragma unroll
      for (int u = 0; u < 2; ++u) {
        int v = tid + 256 * u;
        int row = v >> 2, c8 = (v & 3) * 8;
        ta[u] = *(const v8bf*)(X16 + (size_t)(m0 + row) * C_IN + k0 + c8);
      }
#pragma unroll
      for (int u = 0; u < 2; ++u) {
        int v = tid + 256 * u;
        int row = v >> 4, c8 = (v & 15) * 8;
        tb[u] = *(const v8bf*)(WT16 + (size_t)(k0 + row) * M_OUT + n0 + c8);
      }
#pragma unroll
      for (int u = 0; u < 2; ++u) {
        int v = tid + 256 * u;
        int row = v >> 2, c8 = (v & 3) * 8;
        *(v8bf*)&lA[row * 40 + c8] = ta[u];
      }
#pragma unroll
      for (int u = 0; u < 2; ++u) {
        int v = tid + 256 * u;
        int row = v >> 4, c8 = (v & 15) * 8;
        *(v8bf*)&lB[row * 136 + c8] = tb[u];
      }
    }
    __syncthreads();
    v16bf af = frag_a(&lA[(wave * 16) * 40], 40, lane);
    v16bf bfr[8];
#pragma unroll
    for (int f = 0; f < 8; ++f) bfr[f] = frag_b(&lB[f * 16], 136, lane);
#pragma unroll
    for (int f = 0; f < 8; ++f) acc[f] = wmma_bf16(af, bfr[f], acc[f]);
  }

  // per-row L2 norm over this 128-col (one head) slice
  const int half = lane >> 4;
  float rn[8];
#pragma unroll
  for (int r = 0; r < 8; ++r) {
    float ss = 0.f;
#pragma unroll
    for (int f = 0; f < 8; ++f) ss += acc[f][r] * acc[f][r];
    ss += __shfl_xor(ss, 1); ss += __shfl_xor(ss, 2);
    ss += __shfl_xor(ss, 4); ss += __shfl_xor(ss, 8);
    rn[r] = rsqrtf(ss);
  }

  if (t == 0) {                                   // Q: row-major only
#pragma unroll
    for (int f = 0; f < 8; ++f)
#pragma unroll
      for (int r = 0; r < 8; ++r) {
        int n = m0 + wave * 16 + r + 8 * half;
        int d = f * 16 + (lane & 15);
        Q16[(size_t)h * N_TOK * HD + (size_t)n * HD + d] = (bf16)(acc[f][r] * rn[r]);
      }
  } else if (t == 1) {                            // K: transposed via LDS
    __syncthreads();
#pragma unroll
    for (int f = 0; f < 8; ++f)
#pragma unroll
      for (int r = 0; r < 8; ++r) {
        int nloc = wave * 16 + r + 8 * half;
        int d = f * 16 + (lane & 15);
        lT[d * 136 + nloc] = (bf16)(acc[f][r] * rn[r]);
      }
    __syncthreads();
    v8bf tt[8];
#pragma unroll
    for (int u = 0; u < 8; ++u) {
      int v = tid + 256 * u;
      int d = v >> 4, c8 = (v & 15) * 8;
      tt[u] = *(const v8bf*)&lT[d * 136 + c8];
    }
#pragma unroll
    for (int u = 0; u < 8; ++u) {
      int v = tid + 256 * u;
      int d = v >> 4, c8 = (v & 15) * 8;
      *(v8bf*)(KT16 + (size_t)h * HD * N_TOK + (size_t)d * N_TOK + m0 + c8) = tt[u];
    }
  } else {                                        // V: row-major + transposed
    __syncthreads();
#pragma unroll
    for (int f = 0; f < 8; ++f)
#pragma unroll
      for (int r = 0; r < 8; ++r) {
        int nloc = wave * 16 + r + 8 * half;
        int n = m0 + nloc;
        int d = f * 16 + (lane & 15);
        float val = acc[f][r];
        float nv = val * rn[r];
        size_t di = (size_t)h * N_TOK * HD + (size_t)n * HD + d;
        Vn16[di] = (bf16)nv;
        Vr16[di] = (bf16)val;                                // raw v (bf16)
        out[(size_t)n * 2048 + 1024 + h * HD + d] = val;     // x_ori half
        lT[d * 136 + nloc] = (bf16)nv;
      }
    __syncthreads();
    v8bf tt[8];
#pragma unroll
    for (int u = 0; u < 8; ++u) {
      int v = tid + 256 * u;
      int d = v >> 4, c8 = (v & 15) * 8;
      tt[u] = *(const v8bf*)&lT[d * 136 + c8];
    }
#pragma unroll
    for (int u = 0; u < 8; ++u) {
      int v = tid + 256 * u;
      int d = v >> 4, c8 = (v & 15) * 8;
      *(v8bf*)(VnT16 + (size_t)h * HD * N_TOK + (size_t)d * N_TOK + m0 + c8) = tt[u];
    }
  }
}

// ------------------------------------------- streaming attention, x = A@v
__global__ __launch_bounds__(256) void attn_pass_kern(
    const bf16* __restrict__ Q16, const bf16* __restrict__ KT16,
    const bf16* __restrict__ Vr16, const float* __restrict__ cls,
    float* __restrict__ out, float* __restrict__ Zbuf) {
  __shared__ bf16 lK[128 * 72];        // [d][j] kn^T, stride 72
  __shared__ bf16 lV[64 * 136];        // [j][d] raw v, stride 136
  __shared__ bf16 lP[8 * 16 * 72];     // per-wave probs [16][j], stride 72
  const int tid = threadIdx.x, lane = tid & 31, wave = tid >> 5;
  const int h = blockIdx.y;
  const int rbase = blockIdx.x * 128 + wave * 16;
  const int half = lane >> 4;
  const bf16* Qh  = Q16  + (size_t)h * N_TOK * HD;
  const bf16* KTh = KT16 + (size_t)h * HD * N_TOK;
  const bf16* Vh  = Vr16 + (size_t)h * N_TOK * HD;

  v16bf qf[4];
  {
    const bf16* qp = Qh + (size_t)(rbase + (lane & 15)) * HD + (half << 3);
#pragma unroll
    for (int kk = 0; kk < 4; ++kk)
      qf[kk] = bf_combine(*(const v8bf*)(qp + 32 * kk),
                          *(const v8bf*)(qp + 32 * kk + 16));
  }
  float csi[8];
#pragma unroll
  for (int r = 0; r < 8; ++r) csi[r] = cls[rbase + r + 8 * half] - 0.1f;

  v8f oacc[8] = {};
  float zacc[8] = {0.f, 0.f, 0.f, 0.f, 0.f, 0.f, 0.f, 0.f};
  bf16* lPw = lP + wave * 16 * 72;

  for (int j0 = 0; j0 < N_TOK; j0 += 64) {
    __syncthreads();
    if (j0 + 64 < N_TOK) {
      __builtin_prefetch(KTh + j0 + 64, 0, 1);
      __builtin_prefetch(Vh + (size_t)(j0 + 64) * HD, 0, 1);
    }
    {                                   // stage kn^T + raw V, 8 loads in flight
      v8bf tk[4], tv[4];
#pragma unroll
      for (int u = 0; u < 4; ++u) {
        int v = tid + 256 * u;
        int d = v >> 3, c8 = (v & 7) * 8;
        tk[u] = *(const v8bf*)(KTh + (size_t)d * N_TOK + j0 + c8);
      }
#pragma unroll
      for (int u = 0; u < 4; ++u) {
        int v = tid + 256 * u;
        int jj = v >> 4, d8 = (v & 15) * 8;
        tv[u] = *(const v8bf*)(Vh + (size_t)(j0 + jj) * HD + d8);
      }
#pragma unroll
      for (int u = 0; u < 4; ++u) {
        int v = tid + 256 * u;
        int d = v >> 3, c8 = (v & 7) * 8;
        *(v8bf*)&lK[d * 72 + c8] = tk[u];
      }
#pragma unroll
      for (int u = 0; u < 4; ++u) {
        int v = tid + 256 * u;
        int jj = v >> 4, d8 = (v & 15) * 8;
        *(v8bf*)&lV[jj * 136 + d8] = tv[u];
      }
    }
    __syncthreads();

#pragma unroll
    for (int jt = 0; jt < 4; ++jt) {            // S = qn @ kn^T
      v16bf bk[4];
#pragma unroll
      for (int kk = 0; kk < 4; ++kk)
        bk[kk] = frag_b(&lK[(kk * 32) * 72 + jt * 16], 72, lane);
      v8f s = {};
#pragma unroll
      for (int kk = 0; kk < 4; ++kk) s = wmma_bf16(qf[kk], bk[kk], s);

      int j = j0 + jt * 16 + (lane & 15);
      float csj = cls[j];
#pragma unroll
      for (int r = 0; r < 8; ++r) {
        float m = (csj > csi[r]) ? 1.f : 0.f;
        float p = __expf(s[r] * SCALE_F * csj * m);
        zacc[r] += p;
        lPw[(r + 8 * half) * 72 + jt * 16 + (lane & 15)] = (bf16)p;
      }
    }
#pragma unroll
    for (int kk = 0; kk < 2; ++kk) {            // O += P @ V
      v16bf pf = frag_a(lPw + kk * 32, 72, lane);
      v16bf vfr[8];
#pragma unroll
      for (int n = 0; n < 8; ++n)
        vfr[n] = frag_b(&lV[(kk * 32) * 136 + n * 16], 136, lane);
#pragma unroll
      for (int n = 0; n < 8; ++n) oacc[n] = wmma_bf16(pf, vfr[n], oacc[n]);
    }
  }

#pragma unroll
  for (int r = 0; r < 8; ++r) {                 // Z + 1/Z
    float z = zacc[r];
    z += __shfl_xor(z, 1); z += __shfl_xor(z, 2);
    z += __shfl_xor(z, 4); z += __shfl_xor(z, 8);
    if ((lane & 15) == 0) Zbuf[(size_t)h * N_TOK + rbase + r + 8 * half] = z;
    zacc[r] = 1.f / z;
  }
#pragma unroll
  for (int n = 0; n < 8; ++n)
#pragma unroll
    for (int r = 0; r < 8; ++r) {
      int i = rbase + r + 8 * half;
      int col = h * HD + n * 16 + (lane & 15);
      out[(size_t)i * 2048 + col] = oacc[n][r] * zacc[r];
    }
}

// --------------------- head-averaged attn (SA) + vn@vn^T threshold mask
__global__ __launch_bounds__(256) void sim_pass_kern(
    const bf16* __restrict__ Q16, const bf16* __restrict__ KT16,
    const bf16* __restrict__ Vn16, const bf16* __restrict__ VnT16,
    const float* __restrict__ cls, const float* __restrict__ Zbuf,
    float* __restrict__ simout, float* __restrict__ maskb) {
  __shared__ bf16 lK[128 * 72];        // kn^T  [d][j]
  __shared__ bf16 lW[128 * 72];        // vn^T  [d][j]
  const int tid = threadIdx.x, lane = tid & 31, wave = tid >> 5;
  const int rbase = blockIdx.x * 128 + wave * 16;
  const int j0 = blockIdx.y * 64;
  const int half = lane >> 4;

  float csi[8];
#pragma unroll
  for (int r = 0; r < 8; ++r) csi[r] = cls[rbase + r + 8 * half] - 0.1f;

  v8f sa[4] = {}, sr[4] = {};
  for (int h = 0; h < NH; ++h) {
    const bf16* KTh = KT16 + (size_t)h * HD * N_TOK;
    const bf16* WTh = VnT16 + (size_t)h * HD * N_TOK;
    __syncthreads();
    {                                   // stage kn^T + vn^T, 8 loads in flight
      v8bf tk[4], tw[4];
#pragma unroll
      for (int u = 0; u < 4; ++u) {
        int v = tid + 256 * u;
        int d = v >> 3, c8 = (v & 7) * 8;
        tk[u] = *(const v8bf*)(KTh + (size_t)d * N_TOK + j0 + c8);
        tw[u] = *(const v8bf*)(WTh + (size_t)d * N_TOK + j0 + c8);
      }
#pragma unroll
      for (int u = 0; u < 4; ++u) {
        int v = tid + 256 * u;
        int d = v >> 3, c8 = (v & 7) * 8;
        *(v8bf*)&lK[d * 72 + c8] = tk[u];
        *(v8bf*)&lW[d * 72 + c8] = tw[u];
      }
    }
    __syncthreads();

    v16bf qf[4], vf[4];
    {
      const bf16* qp = Q16 + (size_t)h * N_TOK * HD +
                       (size_t)(rbase + (lane & 15)) * HD + (half << 3);
      const bf16* vp = Vn16 + (size_t)h * N_TOK * HD +
                       (size_t)(rbase + (lane & 15)) * HD + (half << 3);
#pragma unroll
      for (int kk = 0; kk < 4; ++kk) {
        qf[kk] = bf_combine(*(const v8bf*)(qp + 32 * kk),
                            *(const v8bf*)(qp + 32 * kk + 16));
        vf[kk] = bf_combine(*(const v8bf*)(vp + 32 * kk),
                            *(const v8bf*)(vp + 32 * kk + 16));
      }
    }
    float rZ[8];
#pragma unroll
    for (int r = 0; r < 8; ++r)
      rZ[r] = 1.f / Zbuf[(size_t)h * N_TOK + rbase + r + 8 * half];

#pragma unroll
    for (int jt = 0; jt < 4; ++jt) {
      v16bf bk[4], bw[4];
#pragma unroll
      for (int kk = 0; kk < 4; ++kk) {
        bk[kk] = frag_b(&lK[(kk * 32) * 72 + jt * 16], 72, lane);
        bw[kk] = frag_b(&lW[(kk * 32) * 72 + jt * 16], 72, lane);
      }
      v8f s = {}, sv = {};
#pragma unroll
      for (int kk = 0; kk < 4; ++kk) s = wmma_bf16(qf[kk], bk[kk], s);
#pragma unroll
      for (int kk = 0; kk < 4; ++kk) sv = wmma_bf16(vf[kk], bw[kk], sv);

      int j = j0 + jt * 16 + (lane & 15);
      float csj = cls[j];
#pragma unroll
      for (int r = 0; r < 8; ++r) {
        float m = (csj > csi[r]) ? 1.f : 0.f;
        sa[jt][r] += __expf(s[r] * SCALE_F * csj * m) * rZ[r];
        sr[jt][r] += sv[r];
      }
    }
  }
#pragma unroll
  for (int jt = 0; jt < 4; ++jt)
#pragma unroll
    for (int r = 0; r < 8; ++r) {
      int i = rbase + r + 8 * half;
      int j = j0 + jt * 16 + (lane & 15);
      simout[(size_t)i * 2048 + j] = sa[jt][r] * 0.125f;
      maskb[(size_t)i * 2048 + j] = (sr[jt][r] * 0.125f > 0.75f) ? 1.f : 0.f;
    }
}

// -------- fused softmax+mask+renorm: out = mask*exp(SA)/sum(mask*exp(SA))
__global__ __launch_bounds__(256) void sim_norm_kern(float* __restrict__ sim,
                                                     const float* __restrict__ maskb) {
  __shared__ float red[256];
  const int i = blockIdx.x, tid = threadIdx.x;
  float acc = 0.f;
#pragma unroll
  for (int j = tid; j < N_TOK; j += 256) {
    size_t idx = (size_t)i * 2048 + j;
    acc += maskb[idx] * __expf(sim[idx]);
  }
  red[tid] = acc;
  __syncthreads();
  for (int s = 128; s > 0; s >>= 1) {
    if (tid < s) red[tid] += red[tid + s];
    __syncthreads();
  }
  float inv = 1.f / red[0];
#pragma unroll
  for (int j = tid; j < N_TOK; j += 256) {
    size_t idx = (size_t)i * 2048 + j;
    sim[idx] = maskb[idx] * __expf(sim[idx]) * inv;
  }
}

// ---------------------------------------------------------------- launcher
extern "C" void kernel_launch(void* const* d_in, const int* in_sizes, int n_in,
                              void* d_out, int out_size, void* d_ws, size_t ws_size,
                              hipStream_t stream) {
  const float* x   = (const float*)d_in[0];   // [1,2048,1024]
  const float* cls = (const float*)d_in[1];   // [2048]
  // d_in[2] = fg_score (unused by reference path)
  const float* W   = (const float*)d_in[3];   // [3072,1024]
  float* out = (float*)d_out;                           // [2048,2048] x_out
  float* sim = out + (size_t)N_TOK * 2048;              // [2048,2048] sim

  char* w = (char*)d_ws;
  bf16* X16   = (bf16*)w; w += (size_t)N_TOK * C_IN * 2;          // 4 MiB
  bf16* WT16  = (bf16*)w; w += (size_t)C_IN * M_OUT * 2;          // 6 MiB
  bf16* Q16   = (bf16*)w; w += (size_t)NH * N_TOK * HD * 2;       // 4 MiB
  bf16* KT16  = (bf16*)w; w += (size_t)NH * N_TOK * HD * 2;       // 4 MiB
  bf16* Vn16  = (bf16*)w; w += (size_t)NH * N_TOK * HD * 2;       // 4 MiB
  bf16* VnT16 = (bf16*)w; w += (size_t)NH * N_TOK * HD * 2;       // 4 MiB
  bf16* Vr16  = (bf16*)w; w += (size_t)NH * N_TOK * HD * 2;       // 4 MiB
  float* Zbuf  = (float*)w; w += (size_t)NH * N_TOK * 4;          // 64 KiB
  float* maskb = (float*)w; w += (size_t)N_TOK * 2048 * 4;        // 16 MiB

  conv_x_kern<<<(N_TOK * C_IN) / 2048, 256, 0, stream>>>(x, X16);
  dim3 gw(M_OUT / 64, C_IN / 64);      // 48 x 16
  conv_wt_kern<<<gw, 256, 0, stream>>>(W, WT16);

  dim3 g1(N_TOK / 128, M_OUT / 128);   // 16 x 24
  qkv_gemm_kern<<<g1, 256, 0, stream>>>(X16, WT16, Q16, KT16, Vn16, VnT16, Vr16, out);

  dim3 g2(N_TOK / 128, NH);            // 16 x 8
  attn_pass_kern<<<g2, 256, 0, stream>>>(Q16, KT16, Vr16, cls, out, Zbuf);

  dim3 g3(N_TOK / 128, N_TOK / 64);    // 16 x 32
  sim_pass_kern<<<g3, 256, 0, stream>>>(Q16, KT16, Vn16, VnT16, cls, Zbuf, sim, maskb);

  sim_norm_kern<<<N_TOK, 256, 0, stream>>>(sim, maskb);
}
